// Heads_58987080843535
// MI455X (gfx1250) — compile-verified, matrix operand order checked by reading, not run on hardware
//
#include <hip/hip_runtime.h>

// ---------------------------------------------------------------------------
// Multi-head attention, B=4 T=1024 C=1024 H=8, causal.
// Pipeline: cast f32->f16, QKV WMMA GEMM (V stored transposed), causal
// S = Q K^T WMMA GEMM, row softmax -> P (f16, zero-padded), O = P V WMMA GEMM.
// GEMMs: v_wmma_f32_16x16x32_f16, 128x128 block tiles, double-buffered LDS
// staged via global_load_async_to_lds_b128 (ASYNCcnt), one barrier per k-step.
// ---------------------------------------------------------------------------

typedef _Float16 f16x4  __attribute__((ext_vector_type(4)));
typedef _Float16 f16x8  __attribute__((ext_vector_type(8)));
typedef _Float16 f16x16 __attribute__((ext_vector_type(16)));
typedef float    f32x8  __attribute__((ext_vector_type(8)));

#define BT  4096        // B*T rows
#define CC  1024        // channels / head dim
#define HQ  8           // heads
#define HEAD_ELEMS  4194304u   // 4096*1024 elements per head (Q/K/Vt)
#define W_ELEMS     1048576u   // 1024*1024 per head weight
#define S_ELEMS     1048576u   // 1024*1024 per (b,h) score plane
#define LDSL 40                // LDS row stride in halves (32 + 8 pad)
#define TILE_HALVES (128 * LDSL)

// Async DMA: copy 16 bytes per lane from global memory directly into LDS.
// Tracked by ASYNCcnt; completion enforced with s_wait_asynccnt.
__device__ __forceinline__ void async_ld_b128(unsigned lds_byte_off, const _Float16* g)
{
    asm volatile("global_load_async_to_lds_b128 %0, %1, off"
                 :: "v"(lds_byte_off), "v"(g)
                 : "memory");
}

__device__ __forceinline__ void wait_async0()
{
    asm volatile("s_wait_asynccnt 0x0" ::: "memory");
}

// Load a 16x32 f16 WMMA operand fragment (A or B^T style: 16 rows x 32 k,
// row-major, contiguous along k). Per ISA: lanes 0-15 -> k {0..7,16..23},
// lanes 16-31 -> k {8..15,24..31}; row = lane & 15.
__device__ __forceinline__ f16x16 frag_ld(const _Float16* p) {
    f16x8 lo = *(const f16x8*)p;
    f16x8 hi = *(const f16x8*)(p + 16);
    f16x16 r;
#pragma unroll
    for (int i = 0; i < 8; ++i) { r[i] = lo[i]; r[i + 8] = hi[i]; }
    return r;
}

// Core 128x128 block GEMM: D += A(128 x 32*kTiles) * Bt(128 x 32*kTiles)^T.
// Both operands row-major with contiguous k. 256 threads = 8 waves, each wave
// owns a 32x64 sub-tile = 2x4 fragments of 16x16 f32. Double-buffered LDS
// tiles filled by async-to-LDS DMA overlapped with the WMMA compute.
__device__ __forceinline__ void gemm_core(const _Float16* __restrict__ Ag, int lda,
                                          const _Float16* __restrict__ Bg, int ldb,
                                          int kTiles, f32x8 acc[2][4])
{
    __shared__ _Float16 lA[2][TILE_HALVES];
    __shared__ _Float16 lB[2][TILE_HALVES];

    const int tid  = threadIdx.x;
    const int lane = tid & 31;
    const int w    = tid >> 5;
    const int wr   = w >> 1;          // 0..3  (wave row)
    const int wc   = w & 1;           // 0..1  (wave col)
    const int fr   = lane & 15;       // fragment row
    const int kh   = (lane >> 4) * 8; // k-half select

    // Per-thread staging chunks: 2x (A,B) 16B chunks per tile.
    const int r0 = tid >> 2;          // rows 0..63 (plus +64 for chunk 1)
    const int c0 = (tid & 3) * 8;     // k-column within the 32-wide tile

    const unsigned aBase = (unsigned)(size_t)&lA[0][0];
    const unsigned bBase = (unsigned)(size_t)&lB[0][0];

    auto issue_tile = [&](int kt, int buf) {
        const int k0 = kt * 32;
#pragma unroll
        for (int c = 0; c < 2; ++c) {
            const int row = r0 + c * 64;
            const unsigned loff = (unsigned)((buf * TILE_HALVES + row * LDSL + c0) * 2);
            async_ld_b128(aBase + loff, &Ag[(long)row * lda + k0 + c0]);
            async_ld_b128(bBase + loff, &Bg[(long)row * ldb + k0 + c0]);
        }
    };

    issue_tile(0, 0);

    for (int kt = 0; kt < kTiles; ++kt) {
        const int buf = kt & 1;
        wait_async0();          // this wave's copies for tile kt have landed
        __syncthreads();        // every wave's copies visible; prev buf free
        if (kt + 1 < kTiles)
            issue_tile(kt + 1, buf ^ 1);   // DMA next tile under the compute

        f16x16 af[2], bf[4];
#pragma unroll
        for (int mi = 0; mi < 2; ++mi)
            af[mi] = frag_ld(&lA[buf][(wr * 32 + mi * 16 + fr) * LDSL + kh]);
#pragma unroll
        for (int ni = 0; ni < 4; ++ni)
            bf[ni] = frag_ld(&lB[buf][(wc * 64 + ni * 16 + fr) * LDSL + kh]);

#pragma unroll
        for (int mi = 0; mi < 2; ++mi)
#pragma unroll
            for (int ni = 0; ni < 4; ++ni)
                acc[mi][ni] = __builtin_amdgcn_wmma_f32_16x16x32_f16(
                    false, af[mi], false, bf[ni], (short)0, acc[mi][ni], false, false);
    }
}

__device__ __forceinline__ void acc_zero(f32x8 acc[2][4]) {
#pragma unroll
    for (int mi = 0; mi < 2; ++mi)
#pragma unroll
        for (int ni = 0; ni < 4; ++ni)
            acc[mi][ni] = (f32x8){0.f, 0.f, 0.f, 0.f, 0.f, 0.f, 0.f, 0.f};
}

// ---------------------------------------------------------------------------
// f32 -> f16 cast, 4-wide
// ---------------------------------------------------------------------------
__global__ void __launch_bounds__(256)
cast_f16_kernel(const float* __restrict__ src, _Float16* __restrict__ dst, int n4)
{
    int i = blockIdx.x * 256 + threadIdx.x;
    if (i < n4) {
        float4 f = ((const float4*)src)[i];
        f16x4 h;
        h[0] = (_Float16)f.x; h[1] = (_Float16)f.y;
        h[2] = (_Float16)f.z; h[3] = (_Float16)f.w;
        ((f16x4*)dst)[i] = h;
    }
}

// ---------------------------------------------------------------------------
// QKV projection: for mat in {Q,K,V}, head h:  Y = X * W[h]^T
// X: [4096,1024] f16.  W[h]: [d][c] row-major == B^T layout directly.
// Q,K stored [h][m][d]; V stored transposed [h][d][m] (packed b128 stores).
// grid (32, 8, 24) : (mblk, nblk, mat*8+h), 256 threads.
// ---------------------------------------------------------------------------
__global__ void __launch_bounds__(256)
qkv_gemm_kernel(const _Float16* __restrict__ Xh,
                const _Float16* __restrict__ Wqh,
                const _Float16* __restrict__ Wkh,
                const _Float16* __restrict__ Wvh,
                _Float16* __restrict__ Qh,
                _Float16* __restrict__ Kh,
                _Float16* __restrict__ Vt)
{
    const int mb = blockIdx.x, nb = blockIdx.y;
    const int z = blockIdx.z;
    const int mat = z >> 3, h = z & 7;
    const _Float16* W = (mat == 0) ? Wqh : (mat == 1) ? Wkh : Wvh;

    const _Float16* Ag = Xh + (size_t)mb * 128 * CC;
    const _Float16* Bg = W + (size_t)h * W_ELEMS + (size_t)nb * 128 * CC;

    f32x8 acc[2][4];
    acc_zero(acc);
    gemm_core(Ag, CC, Bg, CC, 32, acc);

    const int lane = threadIdx.x & 31;
    const int w = threadIdx.x >> 5, wr = w >> 1, wc = w & 1;
    const int nIn = lane & 15;
    const int mHi = (lane >> 4) * 8;

    if (mat < 2) {
        _Float16* Dst = ((mat == 0) ? Qh : Kh) + (size_t)h * HEAD_ELEMS;
#pragma unroll
        for (int mi = 0; mi < 2; ++mi)
#pragma unroll
            for (int ni = 0; ni < 4; ++ni) {
                int n = nb * 128 + wc * 64 + ni * 16 + nIn;
#pragma unroll
                for (int r = 0; r < 8; ++r) {
                    int m = mb * 128 + wr * 32 + mi * 16 + mHi + r;
                    Dst[(size_t)m * CC + n] = (_Float16)acc[mi][ni][r];
                }
            }
    } else {
        _Float16* Dst = Vt + (size_t)h * HEAD_ELEMS;  // [d][m], ld = 4096
#pragma unroll
        for (int mi = 0; mi < 2; ++mi)
#pragma unroll
            for (int ni = 0; ni < 4; ++ni) {
                int n  = nb * 128 + wc * 64 + ni * 16 + nIn;        // d
                int m0 = mb * 128 + wr * 32 + mi * 16 + mHi;        // m start (8-aligned)
                f16x8 hv;
#pragma unroll
                for (int r = 0; r < 8; ++r) hv[r] = (_Float16)acc[mi][ni][r];
                *(f16x8*)&Dst[(size_t)n * BT + m0] = hv;
            }
    }
}

// ---------------------------------------------------------------------------
// S = (Q K^T) / sqrt(C), causal blocks only (nb <= mb). f32 scratch [z][t][s].
// grid (8, 8, 32) : (mblk, nblk, b*8+h), 256 threads.
// ---------------------------------------------------------------------------
__global__ void __launch_bounds__(256)
scores_kernel(const _Float16* __restrict__ Qh,
              const _Float16* __restrict__ Kh,
              float* __restrict__ S)
{
    const int mb = blockIdx.x, nb = blockIdx.y;
    if (nb > mb) return;            // uniform early-exit above the diagonal
    const int z = blockIdx.z;
    const int b = z >> 3, h = z & 7;

    const _Float16* Ag = Qh + (size_t)h * HEAD_ELEMS + (size_t)(b * 1024 + mb * 128) * CC;
    const _Float16* Bg = Kh + (size_t)h * HEAD_ELEMS + (size_t)(b * 1024 + nb * 128) * CC;

    f32x8 acc[2][4];
    acc_zero(acc);
    gemm_core(Ag, CC, Bg, CC, 32, acc);

    float* Sb = S + (size_t)z * S_ELEMS;
    const float scale = 0.03125f;   // 1/sqrt(1024)

    const int lane = threadIdx.x & 31;
    const int w = threadIdx.x >> 5, wr = w >> 1, wc = w & 1;
    const int nIn = lane & 15;
    const int mHi = (lane >> 4) * 8;
#pragma unroll
    for (int mi = 0; mi < 2; ++mi)
#pragma unroll
        for (int ni = 0; ni < 4; ++ni) {
            int s = nb * 128 + wc * 64 + ni * 16 + nIn;
#pragma unroll
            for (int r = 0; r < 8; ++r) {
                int t = mb * 128 + wr * 32 + mi * 16 + mHi + r;
                Sb[(size_t)t * 1024 + s] = acc[mi][ni][r] * scale;
            }
        }
}

// ---------------------------------------------------------------------------
// Row softmax over S[z][t][0..t], write P f16 with zeros for s > t.
// grid (1024, 32), 256 threads (8 wave32's).
// ---------------------------------------------------------------------------
__global__ void __launch_bounds__(256)
softmax_kernel(const float* __restrict__ S, _Float16* __restrict__ P)
{
    const int t = blockIdx.x;
    const int z = blockIdx.y;
    const float* row = S + (size_t)z * S_ELEMS + (size_t)t * 1024;
    _Float16* prow   = P + (size_t)z * S_ELEMS + (size_t)t * 1024;
    const int len = t + 1;

    const int tid = threadIdx.x;
    const int lane = tid & 31, wid = tid >> 5;
    __shared__ float red[8];

    float m = -3.4e38f;
    for (int s = tid; s < len; s += 256) m = fmaxf(m, row[s]);
#pragma unroll
    for (int off = 16; off; off >>= 1) m = fmaxf(m, __shfl_xor(m, off, 32));
    if (lane == 0) red[wid] = m;
    __syncthreads();
    float M = red[0];
#pragma unroll
    for (int i = 1; i < 8; ++i) M = fmaxf(M, red[i]);
    __syncthreads();

    float sum = 0.f;
    for (int s = tid; s < len; s += 256) sum += __expf(row[s] - M);
#pragma unroll
    for (int off = 16; off; off >>= 1) sum += __shfl_xor(sum, off, 32);
    if (lane == 0) red[wid] = sum;
    __syncthreads();
    float tot = 0.f;
#pragma unroll
    for (int i = 0; i < 8; ++i) tot += red[i];
    const float inv = 1.0f / tot;

    for (int s = tid; s < 1024; s += 256) {
        float v = (s < len) ? __expf(row[s] - M) * inv : 0.0f;
        prow[s] = (_Float16)v;
    }
}

// ---------------------------------------------------------------------------
// O = P * V : A = P[z] rows (k = s), Bt = Vt[h] rows (row = d, k = s contig).
// k-loop truncated at causal boundary (P is zero beyond it anyway).
// Output f32 into [B, T, H*C] layout. grid (8, 8, 32), 256 threads.
// ---------------------------------------------------------------------------
__global__ void __launch_bounds__(256)
pv_gemm_kernel(const _Float16* __restrict__ P,
               const _Float16* __restrict__ Vt,
               float* __restrict__ out)
{
    const int mb = blockIdx.x, db = blockIdx.y;
    const int z = blockIdx.z;
    const int b = z >> 3, h = z & 7;

    const _Float16* Ag = P + (size_t)z * S_ELEMS + (size_t)(mb * 128) * 1024;
    const _Float16* Bg = Vt + (size_t)h * HEAD_ELEMS + (size_t)(db * 128) * BT + (size_t)b * 1024;
    const int kTiles = (mb + 1) * 4;   // covers s <= t for every row in block

    f32x8 acc[2][4];
    acc_zero(acc);
    gemm_core(Ag, 1024, Bg, BT, kTiles, acc);

    const int lane = threadIdx.x & 31;
    const int w = threadIdx.x >> 5, wr = w >> 1, wc = w & 1;
    const int nIn = lane & 15;
    const int mHi = (lane >> 4) * 8;
#pragma unroll
    for (int mi = 0; mi < 2; ++mi)
#pragma unroll
        for (int ni = 0; ni < 4; ++ni) {
            int d = db * 128 + wc * 64 + ni * 16 + nIn;
#pragma unroll
            for (int r = 0; r < 8; ++r) {
                int t = mb * 128 + wr * 32 + mi * 16 + mHi + r;
                out[(size_t)(b * 1024 + t) * (HQ * CC) + (size_t)h * CC + d] = acc[mi][ni][r];
            }
        }
}

// ---------------------------------------------------------------------------
extern "C" void kernel_launch(void* const* d_in, const int* in_sizes, int n_in,
                              void* d_out, int out_size, void* d_ws, size_t ws_size,
                              hipStream_t stream)
{
    (void)in_sizes; (void)n_in; (void)out_size; (void)ws_size;
    const float* data = (const float*)d_in[0];
    const float* Wq   = (const float*)d_in[1];
    const float* Wk   = (const float*)d_in[2];
    const float* Wv   = (const float*)d_in[3];
    float* out = (float*)d_out;

    _Float16* p = (_Float16*)d_ws;
    _Float16* Xh  = p; p += 4194304;               // [4096,1024] f16
    _Float16* Wqh = p; p += 8388608;               // [8,1024,1024] f16
    _Float16* Wkh = p; p += 8388608;
    _Float16* Wvh = p; p += 8388608;
    _Float16* Qh  = p; p += 33554432;              // [8][4096][1024]
    _Float16* Kh  = p; p += 33554432;
    _Float16* Vt  = p; p += 33554432;              // [8][1024][4096] (d-major)
    float* S = (float*)p;                          // [32][1024][1024] f32
    _Float16* Pm = (_Float16*)(S + 33554432);      // [32][1024][1024] f16

    cast_f16_kernel<<<(1048576 + 255) / 256, 256, 0, stream>>>(data, Xh, 1048576);
    cast_f16_kernel<<<(2097152 + 255) / 256, 256, 0, stream>>>(Wq, Wqh, 2097152);
    cast_f16_kernel<<<(2097152 + 255) / 256, 256, 0, stream>>>(Wk, Wkh, 2097152);
    cast_f16_kernel<<<(2097152 + 255) / 256, 256, 0, stream>>>(Wv, Wvh, 2097152);

    dim3 gQKV(32, 8, 24);
    qkv_gemm_kernel<<<gQKV, 256, 0, stream>>>(Xh, Wqh, Wkh, Wvh, Qh, Kh, Vt);

    dim3 gS(8, 8, 32);
    scores_kernel<<<gS, 256, 0, stream>>>(Qh, Kh, S);

    dim3 gSm(1024, 32);
    softmax_kernel<<<gSm, 256, 0, stream>>>(S, Pm);

    dim3 gPV(8, 8, 32);
    pv_gemm_kernel<<<gPV, 256, 0, stream>>>(Pm, Vt, out);
}
